// MultiHeadAttention_65283502899616
// MI455X (gfx1250) — matrix operand order repaired and checked by reading, not compiled, not run on hardware
//
#include <hip/hip_runtime.h>

// ---------------------------------------------------------------------------
// MI455X (gfx1250) MHA forward: bf16 WMMA GEMMs + flash attention,
// with TDM (tensor_load_to_lds) tile staging and async global->LDS copies.
// ---------------------------------------------------------------------------

typedef __attribute__((ext_vector_type(16))) __bf16        bf16x16;
typedef __attribute__((ext_vector_type(8)))  float         floatx8;
typedef __attribute__((ext_vector_type(8)))  unsigned int  uintx8;
typedef __attribute__((ext_vector_type(4)))  unsigned int  uint32x4;
typedef __attribute__((ext_vector_type(4)))  int           int32x4;
typedef __attribute__((ext_vector_type(8)))  int           int32x8;

constexpr int Bn = 4, Sn = 2048, En = 1024, Hn = 16, Dn = 64;
constexpr int BS = Bn * Sn;                 // 8192 rows
constexpr float ATTN_SCALE = 0.125f;        // 1/sqrt(64)

static __device__ __forceinline__ unsigned short f2bf(float f) {
  unsigned int u = __float_as_uint(f);
  u += 0x7FFFu + ((u >> 16) & 1u);          // round-to-nearest-even
  return (unsigned short)(u >> 16);
}

static __device__ __forceinline__ floatx8 wmma_bf16(bf16x16 a, bf16x16 b, floatx8 c) {
  // D = A(16x32 bf16) * B(32x16 bf16) + C(16x16 f32)   [wave32]
  return __builtin_amdgcn_wmma_f32_16x16x32_bf16(false, a, false, b,
                                                 (short)0, c, false, false);
}

// Low 32 bits of a generic pointer to __shared__ = LDS byte offset.
template <typename T>
static __device__ __forceinline__ unsigned lds_off(T* p) {
  return (unsigned)(size_t)p;
}

// ---------------------------------------------------------------------------
// TDM: DMA one [128 rows x 32 cols] bf16 tile (row-major, row stride =
// stride_elems) from global into LDS at lds_byte. Issued by one wave;
// completion tracked with TENSORcnt.
// ---------------------------------------------------------------------------
static __device__ __forceinline__ void tdm_load_tile_128x32(
    unsigned lds_byte, const unsigned short* gptr,
    unsigned tensor_cols, unsigned tensor_rows, unsigned stride_elems) {
  unsigned long long ga = (unsigned long long)(size_t)gptr;
  uint32x4 g0;
  g0[0] = 1u;                                    // count=1 (valid descriptor)
  g0[1] = lds_byte;                              // lds_addr
  g0[2] = (unsigned)ga;                          // global_addr[31:0]
  g0[3] = (unsigned)(ga >> 32) | (2u << 30);     // global_addr[56:32] | type=2
  int32x8 g1;
  g1[0] = (int)(1u << 16);                       // wg_mask=0, data_size=1 (2B)
  g1[1] = (int)((tensor_cols & 0xFFFFu) << 16);  // tensor_dim0[15:0]
  g1[2] = (int)((tensor_cols >> 16) | ((tensor_rows & 0xFFFFu) << 16)); // dim0 hi | dim1 lo
  g1[3] = (int)((tensor_rows >> 16) | (32u << 16));                     // dim1 hi | tile_dim0=32
  g1[4] = 128;                                   // tile_dim1=128, tile_dim2=0
  g1[5] = (int)stride_elems;                     // tensor_dim0_stride[31:0]
  g1[6] = 0;                                     // stride hi | dim1_stride lo
  g1[7] = 0;
  int32x4 z4 = (int32x4){0, 0, 0, 0};
#if __clang_major__ >= 23
  int32x8 z8 = (int32x8){0, 0, 0, 0, 0, 0, 0, 0};
  __builtin_amdgcn_tensor_load_to_lds(g0, g1, z4, z4, z8, 0);
#else
  __builtin_amdgcn_tensor_load_to_lds(g0, g1, z4, z4, 0);
#endif
}

// ---------------------------------------------------------------------------
// fp32 -> bf16 elementwise convert
// ---------------------------------------------------------------------------
__global__ __launch_bounds__(256) void k_f32_to_bf16(
    const float* __restrict__ in, unsigned short* __restrict__ out, int n) {
  int i = blockIdx.x * blockDim.x + threadIdx.x;
  int stride = gridDim.x * blockDim.x;
  for (; i < n; i += stride) out[i] = f2bf(in[i]);
}

// fp32 W[k][n] -> bf16 Wt[n][k]  (so GEMM B-operand is K-contiguous)
__global__ __launch_bounds__(256) void k_transpose_bf16(
    const float* __restrict__ W, unsigned short* __restrict__ Wt, int n) {
  int i = blockIdx.x * blockDim.x + threadIdx.x;
  if (i < n * n) {
    int nn = i / n;
    int kk = i - nn * n;
    Wt[i] = f2bf(W[(size_t)kk * n + nn]);
  }
}

// ---------------------------------------------------------------------------
// Tiled bf16 GEMM: Y[M,N] = A[M,K] * Bt[N,K]^T, f32 accumulate via WMMA.
// Block = 256 threads (8 waves). Tile = 128x128, Kstep = 32.
// TDM double-buffered LDS staging (wave 0 drives the DMA).
// OUT_MODE 0: write bf16 into [B,H,S,D] head-major layout (QKV projections)
// OUT_MODE 1: write f32 row-major [M,N] (final output projection)
// ---------------------------------------------------------------------------
template <int OUT_MODE>
__global__ __launch_bounds__(256) void gemm_bf16_wmma(
    const unsigned short* __restrict__ A,
    const unsigned short* __restrict__ Bt,
    void* __restrict__ outp, int M, int N, int K) {
  __shared__ __align__(16) unsigned short aT[2][128][32];
  __shared__ __align__(16) unsigned short bT[2][128][32];

  const int tid  = threadIdx.x;
  const int wave = tid >> 5;
  const int lane = tid & 31;
  const int half = lane >> 4;
  const int l16  = lane & 15;
  const int m0   = blockIdx.y * 128;
  const int n0   = blockIdx.x * 128;

  floatx8 acc[8];
#pragma unroll
  for (int i = 0; i < 8; ++i)
    acc[i] = (floatx8){0.f, 0.f, 0.f, 0.f, 0.f, 0.f, 0.f, 0.f};

  const int ntiles = K >> 5;
  if (wave == 0) {  // prologue: DMA tile 0 into buffer 0
    tdm_load_tile_128x32(lds_off(&aT[0][0][0]), &A[(size_t)m0 * K], K, M, K);
    tdm_load_tile_128x32(lds_off(&bT[0][0][0]), &Bt[(size_t)n0 * K], K, N, K);
  }

  for (int i = 0; i < ntiles; ++i) {
    const int cur = i & 1;
    const int nxt = cur ^ 1;
    if (wave == 0) {
      if (i + 1 < ntiles) {   // pipeline: issue next tile, wait only for current
        const int k1 = (i + 1) * 32;
        tdm_load_tile_128x32(lds_off(&aT[nxt][0][0]), &A[(size_t)m0 * K + k1], K, M, K);
        tdm_load_tile_128x32(lds_off(&bT[nxt][0][0]), &Bt[(size_t)n0 * K + k1], K, N, K);
        __builtin_amdgcn_s_wait_tensorcnt(2);
      } else {
        __builtin_amdgcn_s_wait_tensorcnt(0);
      }
    }
    __syncthreads();

    // ---- A fragment: 16x32, lane(half,m=l16); k = half*16+e
    uintx8 au;
    const unsigned short* ar = &aT[cur][wave * 16 + l16][0];
#pragma unroll
    for (int p = 0; p < 8; ++p) {
      int kk = half * 16 + 2 * p;
      au[p] = *reinterpret_cast<const unsigned int*>(&ar[kk]);
    }
    bf16x16 af = __builtin_bit_cast(bf16x16, au);

    // ---- 8 WMMA n-tiles per wave (A fragment reused) ----
#pragma unroll
    for (int nb = 0; nb < 8; ++nb) {
      uintx8 bu;
      const unsigned short* br = &bT[cur][nb * 16 + l16][half * 16];  // k = half*16+e
#pragma unroll
      for (int p = 0; p < 8; ++p)
        bu[p] = *reinterpret_cast<const unsigned int*>(&br[2 * p]);
      bf16x16 bfv = __builtin_bit_cast(bf16x16, bu);
      acc[nb] = wmma_bf16(af, bfv, acc[nb]);
    }
    __syncthreads();   // buffer `cur` free for reuse at iteration i+2
  }

  // ---- epilogue: C layout row = half*8+r, col = l16 ----
#pragma unroll
  for (int nb = 0; nb < 8; ++nb) {
#pragma unroll
    for (int r = 0; r < 8; ++r) {
      int row = m0 + wave * 16 + half * 8 + r;
      int col = n0 + nb * 16 + l16;
      float v = acc[nb][r];
      if (OUT_MODE == 0) {
        int b = row >> 11;                 // row / Sn
        int s = row & (Sn - 1);
        int h = col >> 6;                  // col / Dn
        int d = col & (Dn - 1);
        ((unsigned short*)outp)[(((size_t)b * Hn + h) * Sn + s) * Dn + d] = f2bf(v);
      } else {
        ((float*)outp)[(size_t)row * N + col] = v;
      }
    }
  }
}

// ---------------------------------------------------------------------------
// Flash attention: block = (b,h) x 128 query rows, 8 waves x 16 rows.
// Key blocks of 32; K tile staged with async global->LDS (ASYNCcnt),
// V transposed through VGPRs; online softmax; all matmuls via WMMA.
// ---------------------------------------------------------------------------
__global__ __launch_bounds__(256) void attn_fwd(
    const unsigned short* __restrict__ Q,    // [B,H,S,D] bf16
    const unsigned short* __restrict__ Kd,   // [B,H,S,D] bf16
    const unsigned short* __restrict__ Vd,   // [B,H,S,D] bf16
    const int* __restrict__ amask,           // [B,S]
    unsigned short* __restrict__ ctx) {      // [B,S,E] bf16
  __shared__ __align__(16) unsigned short kT[32][64];   // [key][d]
  __shared__ __align__(16) unsigned short vT[64][32];   // [d][key] (transposed)
  __shared__ __align__(4)  unsigned short pT[8][16][32];// per-wave P tile
  __shared__ int am[32];

  const int tid  = threadIdx.x;
  const int wave = tid >> 5;
  const int lane = tid & 31;
  const int half = lane >> 4;
  const int l16  = lane & 15;
  const int q0   = blockIdx.x * 128;
  const int bh   = blockIdx.y;
  const int b    = bh >> 4;                 // / Hn
  const int h    = bh & (Hn - 1);
  const size_t base = (size_t)bh * Sn * Dn;

  // ---- Q A-fragments (16 rows x 64 d = two 32-wide chunks), direct global ----
  const int qrow = q0 + wave * 16 + l16;
  bf16x16 qf[2];
#pragma unroll
  for (int c = 0; c < 2; ++c) {
    const unsigned short* qp = &Q[base + (size_t)qrow * Dn + c * 32];
    uint4 lo = *reinterpret_cast<const uint4*>(&qp[half * 16]);
    uint4 hi = *reinterpret_cast<const uint4*>(&qp[half * 16 + 8]);
    uintx8 u = (uintx8){lo.x, lo.y, lo.z, lo.w, hi.x, hi.y, hi.z, hi.w};
    qf[c] = __builtin_bit_cast(bf16x16, u);
  }

  floatx8 oacc[4];
#pragma unroll
  for (int t = 0; t < 4; ++t)
    oacc[t] = (floatx8){0.f, 0.f, 0.f, 0.f, 0.f, 0.f, 0.f, 0.f};
  float mrow[8], lrow[8];
#pragma unroll
  for (int r = 0; r < 8; ++r) { mrow[r] = -1e30f; lrow[r] = 0.f; }

  const int kb_end = (q0 + 127) >> 5;       // causal: skip blocks past diagonal
  for (int kb = 0; kb <= kb_end; ++kb) {
    // ---- stage K block [32][64] via async global->LDS, V transposed ----
    {
      const int key = tid >> 3;
      const int kk  = (tid & 7) * 8;
      const size_t grow = base + (size_t)(kb * 32 + key) * Dn + kk;
      unsigned ldsk = lds_off(&kT[key][kk]);
      const unsigned short* gk = &Kd[grow];
      asm volatile("global_load_async_to_lds_b128 %0, %1, off"
                   :: "v"(ldsk), "v"(gk) : "memory");
      uint4 v4 = *reinterpret_cast<const uint4*>(&Vd[grow]);
      const unsigned short* pv = reinterpret_cast<const unsigned short*>(&v4);
#pragma unroll
      for (int j = 0; j < 8; ++j) vT[kk + j][key] = pv[j];
    }
    if (tid < 32) am[tid] = amask[b * Sn + kb * 32 + tid];
    asm volatile("s_wait_asynccnt 0" ::: "memory");
    __syncthreads();

    // ---- scores = Q K^T : 2 key sub-tiles x 2 contraction chunks ----
    floatx8 sc[2];
#pragma unroll
    for (int kt = 0; kt < 2; ++kt) {
      sc[kt] = (floatx8){0.f, 0.f, 0.f, 0.f, 0.f, 0.f, 0.f, 0.f};
#pragma unroll
      for (int c = 0; c < 2; ++c) {
        uintx8 bu;
        const unsigned short* kp = &kT[kt * 16 + l16][c * 32 + half * 16];
#pragma unroll
        for (int p = 0; p < 8; ++p)
          bu[p] = *reinterpret_cast<const unsigned int*>(&kp[2 * p]);
        bf16x16 bfv = __builtin_bit_cast(bf16x16, bu);
        sc[kt] = wmma_bf16(qf[c], bfv, sc[kt]);
      }
    }

    // ---- scale + causal & padding mask ----
#pragma unroll
    for (int kt = 0; kt < 2; ++kt) {
      const int kg  = kb * 32 + kt * 16 + l16;
      const int pad = am[kt * 16 + l16];
#pragma unroll
      for (int r = 0; r < 8; ++r) {
        const int qg = q0 + wave * 16 + half * 8 + r;
        float sv = sc[kt][r] * ATTN_SCALE;
        sc[kt][r] = ((kg > qg) || (pad == 0)) ? -1e30f : sv;
      }
    }

    // ---- online softmax (row reductions across 16 lanes of same half) ----
#pragma unroll
    for (int r = 0; r < 8; ++r) {
      float x = fmaxf(sc[0][r], sc[1][r]);
      x = fmaxf(x, __shfl_xor(x, 1, 32));
      x = fmaxf(x, __shfl_xor(x, 2, 32));
      x = fmaxf(x, __shfl_xor(x, 4, 32));
      x = fmaxf(x, __shfl_xor(x, 8, 32));
      const float mnew = fmaxf(mrow[r], x);
      const float p0 = __expf(sc[0][r] - mnew);
      const float p1 = __expf(sc[1][r] - mnew);
      sc[0][r] = p0;
      sc[1][r] = p1;
      float rs = p0 + p1;
      rs += __shfl_xor(rs, 1, 32);
      rs += __shfl_xor(rs, 2, 32);
      rs += __shfl_xor(rs, 4, 32);
      rs += __shfl_xor(rs, 8, 32);
      const float alpha = __expf(mrow[r] - mnew);
      lrow[r] = lrow[r] * alpha + rs;
      mrow[r] = mnew;
#pragma unroll
      for (int t = 0; t < 4; ++t) oacc[t][r] *= alpha;
    }

    // ---- reshape P: C-layout -> A-fragment layout via per-wave LDS tile ----
#pragma unroll
    for (int kt = 0; kt < 2; ++kt)
#pragma unroll
      for (int r = 0; r < 8; ++r)
        pT[wave][half * 8 + r][kt * 16 + l16] = f2bf(sc[kt][r]);
    asm volatile("s_wait_dscnt 0" ::: "memory");  // same-wave LDS store->load

    uintx8 pu;
    const unsigned short* pr = &pT[wave][l16][0];
#pragma unroll
    for (int p = 0; p < 8; ++p) {
      int kk = half * 16 + 2 * p;
      pu[p] = *reinterpret_cast<const unsigned int*>(&pr[kk]);
    }
    bf16x16 pf = __builtin_bit_cast(bf16x16, pu);

    // ---- O += P V  (4 d-tiles, contraction over 32 keys) ----
#pragma unroll
    for (int t = 0; t < 4; ++t) {
      uintx8 vu;
      const unsigned short* vr = &vT[t * 16 + l16][half * 16];
#pragma unroll
      for (int p = 0; p < 8; ++p)
        vu[p] = *reinterpret_cast<const unsigned int*>(&vr[2 * p]);
      bf16x16 vf = __builtin_bit_cast(bf16x16, vu);
      oacc[t] = wmma_bf16(pf, vf, oacc[t]);
    }
    __syncthreads();
  }

  // ---- normalize and write ctx in [B,S,E] layout ----
#pragma unroll
  for (int r = 0; r < 8; ++r) {
    const int qg = q0 + wave * 16 + half * 8 + r;
    const float inv = 1.0f / lrow[r];
    unsigned short* cp = &ctx[((size_t)b * Sn + qg) * En + h * Dn];
#pragma unroll
    for (int t = 0; t < 4; ++t) cp[t * 16 + l16] = f2bf(oacc[t][r] * inv);
  }
}

// ---------------------------------------------------------------------------
// Host-side launch
// ---------------------------------------------------------------------------
extern "C" void kernel_launch(void* const* d_in, const int* in_sizes, int n_in,
                              void* d_out, int out_size, void* d_ws, size_t ws_size,
                              hipStream_t stream) {
  (void)in_sizes; (void)n_in; (void)out_size; (void)ws_size;
  const float* x  = (const float*)d_in[0];
  const int*   m  = (const int*)d_in[1];
  const float* Wq = (const float*)d_in[2];
  const float* Wk = (const float*)d_in[3];
  const float* Wv = (const float*)d_in[4];
  const float* Wo = (const float*)d_in[5];
  float* out = (float*)d_out;

  // bf16 workspace layout (u16 elements), total ~88 MB
  unsigned short* ws = (unsigned short*)d_ws;
  const size_t XE = (size_t)BS * En;   // 8,388,608
  const size_t EE = (size_t)En * En;   // 1,048,576
  unsigned short* xb  = ws;
  unsigned short* wqT = xb  + XE;
  unsigned short* wkT = wqT + EE;
  unsigned short* wvT = wkT + EE;
  unsigned short* wvT2= wvT;           // (alias guard, unused)
  unsigned short* woT = wvT + EE;
  unsigned short* Qd  = woT + EE;
  unsigned short* Kd  = Qd + XE;
  unsigned short* Vd  = Kd + XE;
  unsigned short* ctx = Vd + XE;
  (void)wvT2;

  k_f32_to_bf16<<<4096, 256, 0, stream>>>(x, xb, (int)XE);
  k_transpose_bf16<<<4096, 256, 0, stream>>>(Wq, wqT, En);
  k_transpose_bf16<<<4096, 256, 0, stream>>>(Wk, wkT, En);
  k_transpose_bf16<<<4096, 256, 0, stream>>>(Wv, wvT, En);
  k_transpose_bf16<<<4096, 256, 0, stream>>>(Wo, woT, En);

  dim3 gg(En / 128, BS / 128);   // (8, 64)
  gemm_bf16_wmma<0><<<gg, 256, 0, stream>>>(xb, wqT, Qd, BS, En, En);
  gemm_bf16_wmma<0><<<gg, 256, 0, stream>>>(xb, wkT, Kd, BS, En, En);
  gemm_bf16_wmma<0><<<gg, 256, 0, stream>>>(xb, wvT, Vd, BS, En, En);

  dim3 ga(Sn / 128, Bn * Hn);    // (16, 64)
  attn_fwd<<<ga, 256, 0, stream>>>(Qd, Kd, Vd, m, ctx);

  gemm_bf16_wmma<1><<<gg, 256, 0, stream>>>(ctx, woT, out, BS, En, En);
}
